// SpatialMamba2D_90692529422493
// MI455X (gfx1250) — compile-verified, matrix operand order checked by reading, not run on hardware
//
#include <hip/hip_runtime.h>

typedef __attribute__((ext_vector_type(16))) __bf16 v16bf;
typedef __attribute__((ext_vector_type(8)))  float  v8f;

#define NB     4
#define LSEQ   2304
#define CMOD   256
#define DI     512
#define DSTATE 16

static __device__ __forceinline__ unsigned short f32_to_bf16(float f) {
    unsigned u = __float_as_uint(f);
    unsigned r = u + 0x7FFFu + ((u >> 16) & 1u);
    return (unsigned short)(r >> 16);
}

// ---------------- LayerNorm over channel dim (256) ----------------
__global__ void ln_kernel(const float* __restrict__ x, const float* __restrict__ g,
                          const float* __restrict__ b, float* __restrict__ out) {
    __shared__ float red[256];
    int row = blockIdx.x, t = threadIdx.x;
    float v = x[(size_t)row * 256 + t];
    red[t] = v; __syncthreads();
    for (int s = 128; s > 0; s >>= 1) { if (t < s) red[t] += red[t + s]; __syncthreads(); }
    float mu = red[0] * (1.0f / 256.0f);
    __syncthreads();
    float d = v - mu;
    red[t] = d * d; __syncthreads();
    for (int s = 128; s > 0; s >>= 1) { if (t < s) red[t] += red[t + s]; __syncthreads(); }
    float var = red[0] * (1.0f / 256.0f);
    out[(size_t)row * 256 + t] = d * rsqrtf(var + 1e-5f) * g[t] + b[t];
}

// -------- convert fp32 (NB,L,C) -> bf16, optional per-sequence flip --------
__global__ void cvt_flip_kernel(const float* __restrict__ in, unsigned short* __restrict__ out,
                                int total, int flip) {
    int idx = blockIdx.x * 256 + threadIdx.x;
    if (idx >= total) return;
    int c = idx & 255, t2 = idx >> 8;
    int l = t2 % LSEQ, s = t2 / LSEQ;
    int ls = flip ? (LSEQ - 1 - l) : l;
    out[(size_t)(s * LSEQ + l) * 256 + c] = f32_to_bf16(in[(size_t)(s * LSEQ + ls) * 256 + c]);
}

__global__ void f2bf_kernel(const float* __restrict__ in, unsigned short* __restrict__ out, int n) {
    int i = blockIdx.x * 256 + threadIdx.x;
    if (i < n) out[i] = f32_to_bf16(in[i]);
}

// ---------------- bf16 WMMA GEMM: C[M,N] (+)= A[M,K] @ B[N,K]^T ----------------
// BM=128, BN=64, BK=32, 256 threads (8 waves), each wave owns a 32x32 sub-tile
// (4 x v_wmma_f32_16x16x32_bf16 per K-step). Double-buffered LDS, 1 barrier/step.
// K must be a multiple of 32.

static __device__ __forceinline__ void gemm_load_tiles(
    const unsigned short* __restrict__ A, const unsigned short* __restrict__ Bm,
    int M, int N, int K, int m0, int n0, int k0, int lrow, int lcb,
    uint4& ra0, uint4& ra1, uint4& rb)
{
    ra0 = make_uint4(0u, 0u, 0u, 0u);
    ra1 = make_uint4(0u, 0u, 0u, 0u);
    rb  = make_uint4(0u, 0u, 0u, 0u);
    int gm0 = m0 + lrow, gm1 = gm0 + 64, gn = n0 + lrow;
    const unsigned short* pa0 = A + (size_t)gm0 * K + k0 + lcb * 2;
    const unsigned short* pa1 = A + (size_t)gm1 * K + k0 + lcb * 2;
    const unsigned short* pb  = Bm + (size_t)gn * K + k0 + lcb * 2;
    if (gm0 < M) { ra0 = *(const uint4*)pa0; __builtin_prefetch(pa0 + 64, 0, 0); }
    if (gm1 < M) { ra1 = *(const uint4*)pa1; __builtin_prefetch(pa1 + 64, 0, 0); }
    if (gn  < N) { rb  = *(const uint4*)pb;  __builtin_prefetch(pb  + 64, 0, 0); }
}

static __device__ __forceinline__ void gemm_store_tiles(
    unsigned int* sAb, unsigned int* sBb, int lrow, int lcb,
    const uint4& ra0, const uint4& ra1, const uint4& rb)
{
    int a0 = lrow * 17 + lcb, a1 = (lrow + 64) * 17 + lcb, b0 = lrow * 17 + lcb;
    sAb[a0 + 0] = ra0.x; sAb[a0 + 1] = ra0.y; sAb[a0 + 2] = ra0.z; sAb[a0 + 3] = ra0.w;
    sAb[a1 + 0] = ra1.x; sAb[a1 + 1] = ra1.y; sAb[a1 + 2] = ra1.z; sAb[a1 + 3] = ra1.w;
    sBb[b0 + 0] = rb.x;  sBb[b0 + 1] = rb.y;  sBb[b0 + 2] = rb.z;  sBb[b0 + 3] = rb.w;
}

__global__ void gemm_bf16_wmma(const unsigned short* __restrict__ A,
                               const unsigned short* __restrict__ Bm,
                               float* __restrict__ C,
                               const float* __restrict__ bias,
                               int M, int N, int K, int accum) {
    __shared__ unsigned int sA[2][128 * 17];
    __shared__ unsigned int sB[2][64 * 17];
    int t = threadIdx.x;
    int lane = t & 31, w = t >> 5;
    int m0 = blockIdx.y * 128, n0 = blockIdx.x * 64;
    int mw = (w & 3) * 32, nw = (w >> 2) * 32;

    int lrow = t >> 2;            // 0..63
    int lcb  = (t & 3) * 4;       // dword column base (bf16 col = lcb*2)

    v8f acc00 = {0.f,0.f,0.f,0.f,0.f,0.f,0.f,0.f};
    v8f acc01 = acc00, acc10 = acc00, acc11 = acc00;

    union Frag { unsigned int u[8]; v16bf v; };

    int mrow0 = mw + (lane & 15), mrow1 = mrow0 + 16;
    int nrow0 = nw + (lane & 15), nrow1 = nrow0 + 16;
    int kb    = (lane >> 4) * 4;  // 0 or 4 (dword base per ISA K layout)

    uint4 ra0, ra1, rb;
    gemm_load_tiles(A, Bm, M, N, K, m0, n0, 0, lrow, lcb, ra0, ra1, rb);
    gemm_store_tiles(sA[0], sB[0], lrow, lcb, ra0, ra1, rb);

    int buf = 0;
    for (int k0 = 0; k0 < K; k0 += 32) {
        bool hasNext = (k0 + 32) < K;
        if (hasNext)
            gemm_load_tiles(A, Bm, M, N, K, m0, n0, k0 + 32, lrow, lcb, ra0, ra1, rb);

        __syncthreads();

        const unsigned int* pA = sA[buf];
        const unsigned int* pB = sB[buf];
        Frag fa0, fa1, fb0, fb1;
#pragma unroll
        for (int j = 0; j < 4; ++j) {
            fa0.u[j]     = pA[mrow0 * 17 + kb + j];
            fa0.u[4 + j] = pA[mrow0 * 17 + 8 + kb + j];
            fa1.u[j]     = pA[mrow1 * 17 + kb + j];
            fa1.u[4 + j] = pA[mrow1 * 17 + 8 + kb + j];
            fb0.u[j]     = pB[nrow0 * 17 + kb + j];
            fb0.u[4 + j] = pB[nrow0 * 17 + 8 + kb + j];
            fb1.u[j]     = pB[nrow1 * 17 + kb + j];
            fb1.u[4 + j] = pB[nrow1 * 17 + 8 + kb + j];
        }
        acc00 = __builtin_amdgcn_wmma_f32_16x16x32_bf16(false, fa0.v, false, fb0.v,
                                                        (short)0, acc00, false, false);
        acc01 = __builtin_amdgcn_wmma_f32_16x16x32_bf16(false, fa0.v, false, fb1.v,
                                                        (short)0, acc01, false, false);
        acc10 = __builtin_amdgcn_wmma_f32_16x16x32_bf16(false, fa1.v, false, fb0.v,
                                                        (short)0, acc10, false, false);
        acc11 = __builtin_amdgcn_wmma_f32_16x16x32_bf16(false, fa1.v, false, fb1.v,
                                                        (short)0, acc11, false, false);

        if (hasNext)
            gemm_store_tiles(sA[buf ^ 1], sB[buf ^ 1], lrow, lcb, ra0, ra1, rb);
        buf ^= 1;
    }

    int col0 = n0 + nw + (lane & 15);
    int col1 = col0 + 16;
    int rbase0 = m0 + mw + ((lane >> 4) * 8);
    int rbase1 = rbase0 + 16;
#pragma unroll
    for (int i = 0; i < 8; ++i) {
        int r0 = rbase0 + i, r1 = rbase1 + i;
        if (r0 < M) {
            if (col0 < N) {
                float v = acc00[i];
                if (accum) C[(size_t)r0 * N + col0] += v;
                else       C[(size_t)r0 * N + col0] = v + (bias ? bias[col0] : 0.f);
            }
            if (col1 < N) {
                float v = acc01[i];
                if (accum) C[(size_t)r0 * N + col1] += v;
                else       C[(size_t)r0 * N + col1] = v + (bias ? bias[col1] : 0.f);
            }
        }
        if (r1 < M) {
            if (col0 < N) {
                float v = acc10[i];
                if (accum) C[(size_t)r1 * N + col0] += v;
                else       C[(size_t)r1 * N + col0] = v + (bias ? bias[col0] : 0.f);
            }
            if (col1 < N) {
                float v = acc11[i];
                if (accum) C[(size_t)r1 * N + col1] += v;
                else       C[(size_t)r1 * N + col1] = v + (bias ? bias[col1] : 0.f);
            }
        }
    }
}

// ---------------- causal depthwise conv (k=4) + SiLU ----------------
__global__ void conv_silu_kernel(const float* __restrict__ xz, const float* __restrict__ cw,
                                 const float* __restrict__ cb, float* __restrict__ xc,
                                 unsigned short* __restrict__ xcbf, int total) {
    int idx = blockIdx.x * 256 + threadIdx.x;
    if (idx >= total) return;
    int d = idx & 511, t2 = idx >> 9;
    int l = t2 % LSEQ, s = t2 / LSEQ;
    const float* base = xz + (size_t)s * LSEQ * 1024 + d;
    float acc = cb[d];
#pragma unroll
    for (int j = 0; j < 4; ++j) {
        int ls = l - 3 + j;
        if (ls >= 0) acc += cw[d * 4 + j] * base[(size_t)ls * 1024];
    }
    float sv = acc / (1.f + __expf(-acc));
    size_t o = (size_t)(s * LSEQ + l) * 512 + d;
    xc[o] = sv;
    xcbf[o] = f32_to_bf16(sv);
}

// ---------------- dt = softplus(dbl[:,:16] @ Wdt^T + bdt) ----------------
__global__ void dt_kernel(const float* __restrict__ dbl, const float* __restrict__ Wdt,
                          const float* __restrict__ bdt, float* __restrict__ dtout) {
    __shared__ float sd[16];
    int row = blockIdx.y;
    int d = blockIdx.x * 256 + threadIdx.x;
    if (threadIdx.x < 16) sd[threadIdx.x] = dbl[(size_t)row * 48 + threadIdx.x];
    __syncthreads();
    float acc = bdt[d];
#pragma unroll
    for (int r = 0; r < 16; ++r) acc += sd[r] * Wdt[d * 16 + r];
    float sp = (acc > 20.f) ? acc : log1pf(__expf(acc));
    dtout[(size_t)row * 512 + d] = sp;
}

// ---------------- selective scan + D-skip + SiLU(z) gate ----------------
// One channel per thread; B/C staged through double-buffered LDS (1 barrier/step).
__global__ void scan_kernel(const float* __restrict__ xc, const float* __restrict__ dtb,
                            const float* __restrict__ dbl, const float* __restrict__ xz,
                            const float* __restrict__ Alog, const float* __restrict__ Dvec,
                            unsigned short* __restrict__ ybf, int flip) {
    __shared__ float sBC[2][32];
    int s = blockIdx.y;
    int d = blockIdx.x * 128 + threadIdx.x;
    float Arow[DSTATE], h[DSTATE];
#pragma unroll
    for (int n = 0; n < DSTATE; ++n) {
        Arow[n] = -__expf(Alog[(size_t)d * DSTATE + n]);
        h[n] = 0.f;
    }
    float Dp = Dvec[d];
    const float* xcp = xc  + (size_t)s * LSEQ * 512 + d;
    const float* dtp = dtb + (size_t)s * LSEQ * 512 + d;
    const float* zp  = xz  + (size_t)s * LSEQ * 1024 + 512 + d;
    const float* dr  = dbl + (size_t)s * LSEQ * 48;
    unsigned short* yb = ybf + (size_t)s * LSEQ * 512 + d;

    if (threadIdx.x < 32) sBC[0][threadIdx.x] = dr[16 + threadIdx.x];

    for (int l = 0; l < LSEQ; ++l) {
        __syncthreads();
        int cur = l & 1;
        if ((l + 1) < LSEQ && threadIdx.x < 32)
            sBC[cur ^ 1][threadIdx.x] = dr[(size_t)(l + 1) * 48 + 16 + threadIdx.x];

        float x  = xcp[(size_t)l * 512];
        float dt = dtp[(size_t)l * 512];
        float z  = zp[(size_t)l * 1024];
        float dtx = dt * x;
        float y = 0.f;
#pragma unroll
        for (int n = 0; n < DSTATE; ++n) {
            float dA = __expf(dt * Arow[n]);
            h[n] = dA * h[n] + dtx * sBC[cur][n];
            y += h[n] * sBC[cur][16 + n];
        }
        y += x * Dp;
        y *= z / (1.f + __expf(-z));
        int ol = flip ? (LSEQ - 1 - l) : l;
        yb[(size_t)ol * 512] = f32_to_bf16(y);
    }
}

// ------- Weff[c,d] = sum_i Wpo[c, off+i] * Wout[i, d]  (256x512, bf16 out) -------
__global__ void weff_kernel(const float* __restrict__ Wpo, const float* __restrict__ Wout,
                            unsigned short* __restrict__ weff, int off) {
    int idx = blockIdx.x * 256 + threadIdx.x;   // 256*512 total
    int dd = idx & 511, c = idx >> 9;
    float acc = 0.f;
    for (int i = 0; i < 256; ++i)
        acc += Wpo[(size_t)c * 512 + off + i] * Wout[(size_t)i * 512 + dd];
    weff[(size_t)c * 512 + dd] = f32_to_bf16(acc);
}

extern "C" void kernel_launch(void* const* d_in, const int* in_sizes, int n_in,
                              void* d_out, int out_size, void* d_ws, size_t ws_size,
                              hipStream_t stream) {
    (void)in_sizes; (void)n_in; (void)out_size; (void)ws_size;
    const float* x     = (const float*)d_in[0];
    const float* gamma = (const float*)d_in[1];
    const float* beta  = (const float*)d_in[2];
    const float* Wpo   = (const float*)d_in[21];
    const float* bpo   = (const float*)d_in[22];

    const int Mtot = NB * LSEQ;   // 9216

    size_t off = 0;
    auto alloc = [&](size_t bytes) -> void* {
        void* p = (char*)d_ws + off;
        off += (bytes + 255) & ~(size_t)255;
        return p;
    };
    float*          xn    = (float*)alloc((size_t)Mtot * CMOD * 4);
    unsigned short* xnbf  = (unsigned short*)alloc((size_t)Mtot * CMOD * 2);
    unsigned short* winbf = (unsigned short*)alloc((size_t)2 * DI * CMOD * 2);
    unsigned short* wxbf  = (unsigned short*)alloc((size_t)48 * DI * 2);
    float*          xz    = (float*)alloc((size_t)Mtot * 2 * DI * 4);
    float*          xc    = (float*)alloc((size_t)Mtot * DI * 4);
    unsigned short* xcbf  = (unsigned short*)alloc((size_t)Mtot * DI * 2);
    float*          dblb  = (float*)alloc((size_t)Mtot * 48 * 4);
    float*          dtb   = (float*)alloc((size_t)Mtot * DI * 4);
    unsigned short* ybf0  = (unsigned short*)alloc((size_t)Mtot * DI * 2);
    unsigned short* ybf1  = (unsigned short*)alloc((size_t)Mtot * DI * 2);
    unsigned short* weff0 = (unsigned short*)alloc((size_t)CMOD * DI * 2);
    unsigned short* weff1 = (unsigned short*)alloc((size_t)CMOD * DI * 2);
    unsigned short* ybf[2]  = { ybf0, ybf1 };
    unsigned short* weff[2] = { weff0, weff1 };

    ln_kernel<<<Mtot, 256, 0, stream>>>(x, gamma, beta, xn);

    for (int dir = 0; dir < 2; ++dir) {
        const float* Win   = (const float*)d_in[3 + 9 * dir];
        const float* convw = (const float*)d_in[4 + 9 * dir];
        const float* convb = (const float*)d_in[5 + 9 * dir];
        const float* Wx    = (const float*)d_in[6 + 9 * dir];
        const float* Wdt   = (const float*)d_in[7 + 9 * dir];
        const float* bdt   = (const float*)d_in[8 + 9 * dir];
        const float* Alog  = (const float*)d_in[9 + 9 * dir];
        const float* Dv    = (const float*)d_in[10 + 9 * dir];
        const float* Wout  = (const float*)d_in[11 + 9 * dir];

        cvt_flip_kernel<<<(Mtot * CMOD) / 256, 256, 0, stream>>>(xn, xnbf, Mtot * CMOD, dir);
        f2bf_kernel<<<(2 * DI * CMOD + 255) / 256, 256, 0, stream>>>(Win, winbf, 2 * DI * CMOD);
        {
            dim3 g((2 * DI + 63) / 64, (Mtot + 127) / 128);
            gemm_bf16_wmma<<<g, 256, 0, stream>>>(xnbf, winbf, xz, nullptr, Mtot, 2 * DI, CMOD, 0);
        }
        conv_silu_kernel<<<(Mtot * DI) / 256, 256, 0, stream>>>(xz, convw, convb, xc, xcbf, Mtot * DI);
        f2bf_kernel<<<(48 * DI + 255) / 256, 256, 0, stream>>>(Wx, wxbf, 48 * DI);
        {
            dim3 g((48 + 63) / 64, (Mtot + 127) / 128);
            gemm_bf16_wmma<<<g, 256, 0, stream>>>(xcbf, wxbf, dblb, nullptr, Mtot, 48, DI, 0);
        }
        {
            dim3 g(2, Mtot);
            dt_kernel<<<g, 256, 0, stream>>>(dblb, Wdt, bdt, dtb);
        }
        {
            dim3 g(DI / 128, NB);
            scan_kernel<<<g, 128, 0, stream>>>(xc, dtb, dblb, xz, Alog, Dv, ybf[dir], dir);
        }
        weff_kernel<<<(CMOD * DI) / 256, 256, 0, stream>>>(Wpo, Wout, weff[dir], dir * CMOD);
    }

    {
        dim3 g((CMOD + 63) / 64, (Mtot + 127) / 128);
        gemm_bf16_wmma<<<g, 256, 0, stream>>>(ybf[0], weff[0], (float*)d_out, bpo, Mtot, CMOD, DI, 0);
        gemm_bf16_wmma<<<g, 256, 0, stream>>>(ybf[1], weff[1], (float*)d_out, nullptr, Mtot, CMOD, DI, 1);
    }
}
